// AttentionLayer_87050397155952
// MI455X (gfx1250) — compile-verified, matrix operand order checked by reading, not run on hardware
//
#include <hip/hip_runtime.h>
#include <hip/hip_bf16.h>
#include <math.h>

// Problem constants (from reference setup_inputs)
#define B_  64
#define L1_ 512
#define L2_ 512
#define H_  1024
#define A_  1024

typedef __attribute__((ext_vector_type(16))) __bf16 v16bf;
typedef __attribute__((ext_vector_type(8)))  __bf16 v8bf;
typedef __attribute__((ext_vector_type(8)))  float  v8f;

// ---------------------------------------------------------------------------
// Branch-free fast tanh: tanh(x) = copysign(1 - 2/(e^{2|x|}+1), x).
// Saturates naturally (e^{2|x|} -> inf => term -> 0). One v_exp_f32 + one
// v_rcp_f32, no exec-mask branching (unlike libm tanhf).
// ---------------------------------------------------------------------------
__device__ __forceinline__ float fast_tanh(float x) {
    float ax = __builtin_fabsf(x);
    float e  = __expf(ax + ax);
    float t  = 1.0f - 2.0f * __builtin_amdgcn_rcpf(e + 1.0f);
    return __builtin_copysignf(t, x);
}

// ---------------------------------------------------------------------------
// Fragment loaders for V_WMMA_F32_16X16X32_BF16 (wave32).
// ISA 16-bit A-matrix 16x32 layout: lane (l&15) = row, per-lane halves:
//   lanes 0-15 : K = 0..7   (v[0..7]),  K = 16..23 (v[8..15])
//   lanes 16-31: K = 8..15  (v[0..7]),  K = 24..31 (v[8..15])
// i.e. two contiguous 8-element (16B) chunks at koff and koff+16.
// B fragment (32x16, K x N) is symmetric with N striped across lanes.
// ---------------------------------------------------------------------------
__device__ __forceinline__ v16bf frag_from_bf16(const __bf16* __restrict__ base, int ld) {
    const int lane = threadIdx.x & 31;
    const __bf16* p = base + (size_t)(lane & 15) * ld + ((lane >> 4) << 3);
    v8bf lo = *(const v8bf*)(p);
    v8bf hi = *(const v8bf*)(p + 16);
    v16bf r;
#pragma unroll
    for (int i = 0; i < 8; ++i) { r[i] = lo[i]; r[i + 8] = hi[i]; }
    return r;
}

__device__ __forceinline__ v16bf frag_from_f32(const float* __restrict__ base, int ld) {
    const int lane = threadIdx.x & 31;
    const float* p = base + (size_t)(lane & 15) * ld + ((lane >> 4) << 3);
    v16bf r;
#pragma unroll
    for (int i = 0; i < 8; ++i) {
        r[i]     = (__bf16)p[i];
        r[i + 8] = (__bf16)p[16 + i];
    }
    return r;
}

// ---------------------------------------------------------------------------
// Stage 0: Wt[a][h] = bf16(W[h][a])  (4 MB read, once; negligible)
// ---------------------------------------------------------------------------
__global__ void transpose_w_kernel(const float* __restrict__ W, __bf16* __restrict__ Wt) {
    int idx = blockIdx.x * blockDim.x + threadIdx.x;   // over A_*H_
    int a = idx >> 10;          // H_ == 1024
    int h = idx & 1023;
    Wt[idx] = (__bf16)W[(size_t)h * A_ + a];
}

// ---------------------------------------------------------------------------
// Stage 1: P[m][n] = bf16( tanh( sum_k X[m][k]*W[k][n] ) * (diag ? diag[n]:1) )
// Block = 128 threads = 4 waves arranged 2x2; each wave owns a 64x64 tile of
// the 128x128 block tile: 4 A-frags x 4 B-frags -> 16 WMMAs per K-step with
// 4x reuse of every fragment (1.5 b128 loads per WMMA).
// ---------------------------------------------------------------------------
__device__ __forceinline__ void store_proj_tile(const v8f& c, int mbase, int n,
                                                const float* __restrict__ diag,
                                                __bf16* __restrict__ P) {
    const float scale = diag ? diag[n] : 1.0f;   // diagonal_W applied AFTER tanh
#pragma unroll
    for (int i = 0; i < 8; ++i) {
        float v = fast_tanh(c[i]) * scale;
        P[(size_t)(mbase + i) * A_ + n] = (__bf16)v;
    }
}

__global__ __launch_bounds__(128) void proj_tanh_kernel(
    const float* __restrict__ X,      // [M, H_] row-major (f32)
    const __bf16* __restrict__ Wt,    // [A_, H_] row-major (W transposed, bf16)
    const float* __restrict__ diag,   // [A_] or nullptr
    __bf16* __restrict__ P)           // [M, A_] bf16 out
{
    const int wave = threadIdx.x >> 5;
    const int m0 = blockIdx.y * 128 + (wave >> 1) * 64;   // 2 waves stacked in M
    const int n0 = blockIdx.x * 128 + (wave & 1) * 64;    // 2 waves stacked in N

    v8f acc[4][4] = {};

    for (int k = 0; k < H_; k += 32) {
        v16bf a[4];
        v16bf b[4];
#pragma unroll
        for (int i = 0; i < 4; ++i)
            a[i] = frag_from_f32(X + (size_t)(m0 + 16 * i) * H_ + k, H_);
#pragma unroll
        for (int j = 0; j < 4; ++j)
            b[j] = frag_from_bf16(Wt + (size_t)(n0 + 16 * j) * H_ + k, H_);
#pragma unroll
        for (int i = 0; i < 4; ++i)
#pragma unroll
            for (int j = 0; j < 4; ++j)
                acc[i][j] = __builtin_amdgcn_wmma_f32_16x16x32_bf16(
                    false, a[i], false, b[j], (short)0, acc[i][j], false, false);
    }

    // C/D layout: VGPR i, lanes 0-15 -> (M=i, N=lane); lanes 16-31 -> (M=8+i, N=lane-16)
    const int lane = threadIdx.x & 31;
    const int nn = lane & 15;
    const int mm = (lane >> 4) * 8;
#pragma unroll
    for (int i = 0; i < 4; ++i)
#pragma unroll
        for (int j = 0; j < 4; ++j)
            store_proj_tile(acc[i][j], m0 + 16 * i + mm, n0 + 16 * j + nn, diag, P);
}

// ---------------------------------------------------------------------------
// Stage 2: fused  logits = P_lt · P_rt^T  (over A_),  mask, softmax, mask, out.
// One block (256 thr / 8 waves) per (batch, 32-row tile). Wave w computes the
// 32x64 logit slab for columns [64w, 64w+64): 2 A-frags x 4 B-frags -> 8
// WMMAs/K-step. Logits parked in LDS (32 x 516 f32 = 66 KB, fine on a 320 KB
// WGP), then each wave runs a wave32 shuffle-reduction softmax on 4 rows.
// Both operands are row-major over the contraction dim -> contiguous frags.
// ---------------------------------------------------------------------------
__global__ __launch_bounds__(256) void attn_softmax_kernel(
    const __bf16* __restrict__ Plt,     // [B_*L1_, A_]
    const __bf16* __restrict__ Prt,     // [B_*L2_, A_]
    const float* __restrict__ mask_lt,  // [B_*L1_]
    const float* __restrict__ mask_rt,  // [B_*L2_]
    float* __restrict__ out)            // [B_, L1_, L2_]
{
    __shared__ float smem[32][L2_ + 4];

    const int b    = blockIdx.x >> 4;          // 16 row-tiles per batch (L1_/32)
    const int m0   = (blockIdx.x & 15) * 32;
    const int wave = threadIdx.x >> 5;
    const int lane = threadIdx.x & 31;
    const int n0   = wave * 64;

    const __bf16* Abase = Plt + ((size_t)b * L1_ + m0) * A_;
    const __bf16* Bbase = Prt + (size_t)b * L2_ * A_;

    v8f acc[2][4] = {};

    for (int k = 0; k < A_; k += 32) {
        v16bf a0 = frag_from_bf16(Abase + k, A_);
        v16bf a1 = frag_from_bf16(Abase + (size_t)16 * A_ + k, A_);
#pragma unroll
        for (int j = 0; j < 4; ++j) {
            v16bf bb = frag_from_bf16(Bbase + (size_t)(n0 + 16 * j) * A_ + k, A_);
            acc[0][j] = __builtin_amdgcn_wmma_f32_16x16x32_bf16(false, a0, false, bb, (short)0, acc[0][j], false, false);
            acc[1][j] = __builtin_amdgcn_wmma_f32_16x16x32_bf16(false, a1, false, bb, (short)0, acc[1][j], false, false);
        }
    }

    // Spill logits to LDS
#pragma unroll
    for (int mi = 0; mi < 2; ++mi) {
#pragma unroll
        for (int j = 0; j < 4; ++j) {
            const int col = n0 + 16 * j + (lane & 15);
            const int rb  = mi * 16 + (lane >> 4) * 8;
#pragma unroll
            for (int i = 0; i < 8; ++i) smem[rb + i][col] = acc[mi][j][i];
        }
    }
    __syncthreads();

    // Softmax: wave w handles rows 4w..4w+3; 512 cols = 16 per lane.
    const float* mrt = mask_rt + (size_t)b * L2_;
#pragma unroll
    for (int rsel = 0; rsel < 4; ++rsel) {
        const int rr = wave * 4 + rsel;
        const float mlt = mask_lt[(size_t)b * L1_ + m0 + rr];

        float x[16];
        float mrtv[16];
        float mx = -3.402823466e38f;
#pragma unroll
        for (int i = 0; i < 16; ++i) {
            const int c = lane + 32 * i;
            mrtv[i] = mrt[c];
            x[i] = smem[rr][c] * mlt * mrtv[i];   // mask before softmax
            mx = fmaxf(mx, x[i]);
        }
#pragma unroll
        for (int off = 16; off > 0; off >>= 1) mx = fmaxf(mx, __shfl_xor(mx, off, 32));

        float s = 0.0f;
#pragma unroll
        for (int i = 0; i < 16; ++i) { x[i] = __expf(x[i] - mx); s += x[i]; }
#pragma unroll
        for (int off = 16; off > 0; off >>= 1) s += __shfl_xor(s, off, 32);
        const float inv = __builtin_amdgcn_rcpf(s);

        float* o = out + ((size_t)b * L1_ + m0 + rr) * L2_;
#pragma unroll
        for (int i = 0; i < 16; ++i) {
            const int c = lane + 32 * i;
            o[c] = x[i] * inv * mlt * mrtv[i];    // mask again after softmax
        }
    }
}

// ---------------------------------------------------------------------------
// Host-side launch
// ---------------------------------------------------------------------------
extern "C" void kernel_launch(void* const* d_in, const int* in_sizes, int n_in,
                              void* d_out, int out_size, void* d_ws, size_t ws_size,
                              hipStream_t stream) {
    const float* reps_lt = (const float*)d_in[0];   // [B,L1,H]
    const float* reps_rt = (const float*)d_in[1];   // [B,L2,H]
    const float* mask_lt = (const float*)d_in[2];   // [B,L1]
    const float* mask_rt = (const float*)d_in[3];   // [B,L2]
    const float* attn_w1 = (const float*)d_in[4];   // [H,A]
    const float* diagW   = (const float*)d_in[5];   // [1,1,A]
    float* out = (float*)d_out;

    // Workspace layout (bf16): Wt (2MB) | Plt (64MB) | Prt (64MB)
    char* ws = (char*)d_ws;
    __bf16* Wt  = (__bf16*)ws;
    __bf16* Plt = (__bf16*)(ws + (size_t)A_ * H_ * sizeof(__bf16));
    __bf16* Prt = (__bf16*)(ws + (size_t)A_ * H_ * sizeof(__bf16)
                               + (size_t)B_ * L1_ * A_ * sizeof(__bf16));

    // Stage 0: transpose+convert W -> Wt[A][H] bf16
    transpose_w_kernel<<<(A_ * H_) / 256, 256, 0, stream>>>(attn_w1, Wt);

    // Stage 1: projections (tanh; lt side also scaled by diagonal_W)
    dim3 g1(A_ / 128, (B_ * L1_) / 128);
    proj_tanh_kernel<<<g1, 128, 0, stream>>>(reps_lt, Wt, diagW, Plt);
    proj_tanh_kernel<<<g1, 128, 0, stream>>>(reps_rt, Wt, nullptr, Prt);

    // Stage 2: fused batched matmul + masked softmax
    attn_softmax_kernel<<<B_ * (L1_ / 32), 256, 0, stream>>>(Plt, Prt, mask_lt, mask_rt, out);
}
